// Decoder_919123001371
// MI455X (gfx1250) — matrix operand order repaired and checked by reading, not compiled, not run on hardware
//
#include <hip/hip_runtime.h>
#include <math.h>

typedef _Float16 h16;
typedef __attribute__((ext_vector_type(16))) _Float16 v16h;
typedef __attribute__((ext_vector_type(8)))  float    v8f;

#define B_TOT   64
#define T_ENC   400
#define T_DEC   600
#define MEL_N   80
#define ENC_D   256
#define RNN     256
#define PRE1_N  256
#define PRE2_N  128
#define NTHR    1024   // decoder block: 32 waves; waves 0-15 own GEMM N-tiles

// ---------------------------------------------------------------------------
// helpers
// ---------------------------------------------------------------------------
__device__ __forceinline__ v8f z8() {
  v8f v;
#pragma unroll
  for (int i = 0; i < 8; ++i) v[i] = 0.0f;
  return v;
}

__device__ __forceinline__ float sigm(float x) { return 1.0f / (1.0f + __expf(-x)); }

// tanh(x) = 1 - 2/(exp(2x)+1); exact limits at +/-inf, one transcendental.
__device__ __forceinline__ float fast_tanh(float x) {
  float e = __expf(2.0f * x);
  return 1.0f - 2.0f / (e + 1.0f);
}

__device__ __forceinline__ float wred_max(float v) {
#pragma unroll
  for (int off = 16; off; off >>= 1) v = fmaxf(v, __shfl_xor(v, off, 32));
  return v;
}
__device__ __forceinline__ float wred_sum(float v) {
#pragma unroll
  for (int off = 16; off; off >>= 1) v += __shfl_xor(v, off, 32);
  return v;
}

// One 16x16 output tile, accumulating A[16 x 32*ksteps] (LDS, row-major, lda)
// times W^T where W is row-major [N_out x ldw] f16 and rows (n0..n0+15) of W
// starting at column bk0 are used.  Fragment k order used here (contiguous, same for A and B):
//  A: lane m = lane&15; lanes 0-15 hold K 0..15, lanes 16-31 hold K 16..31
//  B: lane n = lane&15; lanes 0-15 hold K 0..15, lanes 16-31 hold K 16..31
__device__ __forceinline__ v8f wmma_acc_seg(v8f acc, const h16* A, int lda,
                                            const h16* W, int ldw,
                                            int n0, int bk0, int ksteps) {
  const int lane = threadIdx.x & 31;
  const int hsel = lane >> 4;
  const int mr = lane & 15;
  const h16* arow = A + mr * lda;
  const h16* wrow = W + (size_t)(n0 + mr) * (size_t)ldw + bk0;
  const int ka = hsel * 8;
  const int kb = hsel * 16;
  for (int ks = 0; ks < ksteps; ++ks) {
    const int k = ks * 32;
    v16h a, b;
#pragma unroll
    for (int j = 0; j < 16; ++j) a[j] = arow[k + kb + j];
#pragma unroll
    for (int j = 0; j < 16; ++j) b[j] = wrow[k + kb + j];
    acc = __builtin_amdgcn_wmma_f32_16x16x32_f16(false, a, false, b,
                                                 (short)0, acc, false, false);
  }
  return acc;
}

// GRU gate math for one wave's 16-column tile (j = col base).
__device__ __forceinline__ void gru_tile(
    int j,
    const h16* X0, int lda0, int ks0,
    const h16* X1, int lda1, int ks1, int bk1,
    const h16* Wih, int ldih,
    const h16* H, const h16* Whh,
    const float* bih, const float* bhh,
    float hnew[8]) {
  v8f ir = z8(), iz = z8(), in_ = z8(), hr = z8(), hz = z8(), hn = z8();
  ir  = wmma_acc_seg(ir,  X0, lda0, Wih, ldih, j,           0, ks0);
  iz  = wmma_acc_seg(iz,  X0, lda0, Wih, ldih, RNN + j,     0, ks0);
  in_ = wmma_acc_seg(in_, X0, lda0, Wih, ldih, 2 * RNN + j, 0, ks0);
  if (X1) {
    ir  = wmma_acc_seg(ir,  X1, lda1, Wih, ldih, j,           bk1, ks1);
    iz  = wmma_acc_seg(iz,  X1, lda1, Wih, ldih, RNN + j,     bk1, ks1);
    in_ = wmma_acc_seg(in_, X1, lda1, Wih, ldih, 2 * RNN + j, bk1, ks1);
  }
  hr = wmma_acc_seg(hr, H, RNN, Whh, RNN, j,           0, 8);
  hz = wmma_acc_seg(hz, H, RNN, Whh, RNN, RNN + j,     0, 8);
  hn = wmma_acc_seg(hn, H, RNN, Whh, RNN, 2 * RNN + j, 0, 8);
  const int lane = threadIdx.x & 31;
  const int col = j + (lane & 15);
  const float bir = bih[col], biz = bih[RNN + col], bin = bih[2 * RNN + col];
  const float bhr = bhh[col], bhz = bhh[RNN + col], bhn = bhh[2 * RNN + col];
#pragma unroll
  for (int r = 0; r < 8; ++r) {
    const int m = r + ((lane >> 4) << 3);
    const float rr = sigm(ir[r] + bir + hr[r] + bhr);
    const float zz = sigm(iz[r] + biz + hz[r] + bhz);
    const float nn = fast_tanh(in_[r] + bin + rr * (hn[r] + bhn));
    const float ho = (float)H[m * RNN + col];
    hnew[r] = (1.0f - zz) * nn + zz * ho;
  }
}

// ---------------------------------------------------------------------------
// fp32 -> f16 weight conversion with K zero-padding ([N][Ksrc] -> [N][Kpad])
// ---------------------------------------------------------------------------
__global__ void convert_w_kernel(const float* __restrict__ src, h16* __restrict__ dst,
                                 int nout, int ksrc, int kpad) {
  int idx = blockIdx.x * blockDim.x + threadIdx.x;
  if (idx >= nout * kpad) return;
  int n = idx / kpad, k = idx - n * kpad;
  dst[idx] = (h16)((k < ksrc) ? src[(size_t)n * ksrc + k] : 0.0f);
}

// ---------------------------------------------------------------------------
// processed_memory = enc @ mem_w^T  (f32 out), WMMA GEMM, 16-row tiles
// ---------------------------------------------------------------------------
__global__ void __launch_bounds__(512) pm_kernel(const float* __restrict__ enc,
                                                 const h16* __restrict__ Wm,
                                                 float* __restrict__ pm) {
  __shared__ h16 sA[16 * ENC_D];
  const size_t row0 = (size_t)blockIdx.x * 16;
  const int tid = threadIdx.x;
  for (int i = tid; i < 16 * ENC_D; i += 512) {
    int m = i >> 8, c = i & 255;
    sA[i] = (h16)enc[(row0 + m) * ENC_D + c];
  }
  __syncthreads();
  const int wv = tid >> 5, lane = tid & 31;
  v8f acc = z8();
  acc = wmma_acc_seg(acc, sA, ENC_D, Wm, ENC_D, wv * 16, 0, 8);
  const int col = wv * 16 + (lane & 15);
#pragma unroll
  for (int r = 0; r < 8; ++r) {
    int m = r + ((lane >> 4) << 3);
    pm[(row0 + m) * ENC_D + col] = acc[r];
  }
}

// ---------------------------------------------------------------------------
// persistent decoder: 1 block per 16 batch rows, 1024 threads = 32 waves.
// Waves 0-15 own the 16 GEMM N-tiles; all 32 waves share the VALU phases.
// ---------------------------------------------------------------------------
struct SMem {
  h16 h_att[16 * RNN];
  h16 h1[16 * RNN];
  h16 h2[16 * RNN];
  h16 ctx[16 * ENC_D];
  h16 pre1[16 * PRE1_N];
  h16 x2[16 * PRE2_N];
  h16 cur[16 * 96];
  h16 df16[16 * RNN];
  float q[16 * RNN];
  float d[16 * RNN];
  float scores[16 * T_ENC];
  float vvec[256];
  int mlen[16];
};

__global__ void __launch_bounds__(NTHR) decoder_kernel(
    const float* __restrict__ enc, const float* __restrict__ inputs,
    const int* __restrict__ memlen,
    const float* __restrict__ b1, const float* __restrict__ b2,
    const float* __restrict__ att_bih, const float* __restrict__ att_bhh,
    const float* __restrict__ vw, const float* __restrict__ proj_b,
    const float* __restrict__ d0_bih, const float* __restrict__ d0_bhh,
    const float* __restrict__ d1_bih, const float* __restrict__ d1_bhh,
    const float* __restrict__ mel_b, const float* __restrict__ stop_w,
    const float* __restrict__ stop_b,
    const float* __restrict__ pm,
    const h16* __restrict__ W1p, const h16* __restrict__ W2,
    const h16* __restrict__ Wq, const h16* __restrict__ WihA,
    const h16* __restrict__ WhhA, const h16* __restrict__ Wproj,
    const h16* __restrict__ Wih0, const h16* __restrict__ Whh0,
    const h16* __restrict__ Wih1, const h16* __restrict__ Whh1,
    const h16* __restrict__ Wmel,
    float* __restrict__ out_mel, float* __restrict__ out_stop,
    float* __restrict__ out_align) {
  extern __shared__ char smem_raw[];
  SMem& S = *reinterpret_cast<SMem*>(smem_raw);

  const int tid = threadIdx.x;
  const int wv = tid >> 5;
  const int lane = tid & 31;
  const int b0 = blockIdx.x * 16;

  for (int i = tid; i < 16 * RNN; i += NTHR) {
    S.h_att[i] = (h16)0.0f; S.h1[i] = (h16)0.0f;
    S.h2[i] = (h16)0.0f;    S.ctx[i] = (h16)0.0f;
  }
  if (tid < 256) S.vvec[tid] = vw[tid];
  if (tid < 16) S.mlen[tid] = memlen[b0 + tid];
  __syncthreads();

  for (int t = 0; t < T_DEC; ++t) {
    // ---- phase 0: shifted input frame -> cur (f16, K padded 80->96) ----
    for (int i = tid; i < 16 * 96; i += NTHR) {
      int m = i / 96, k = i - m * 96;
      float v = 0.0f;
      if (t > 0 && k < MEL_N)
        v = inputs[((size_t)(b0 + m) * T_DEC + (t - 1)) * MEL_N + k];
      S.cur[i] = (h16)v;
    }
    __syncthreads();

    // ---- phase 1: prenet layer 1 (K=96, N=256) ----
    if (wv < 16) {
      const int j = wv * 16;
      v8f acc = z8();
      acc = wmma_acc_seg(acc, S.cur, 96, W1p, 96, j, 0, 3);
      const int col = j + (lane & 15);
      const float bb = b1[col];
#pragma unroll
      for (int r = 0; r < 8; ++r) {
        int m = r + ((lane >> 4) << 3);
        S.pre1[m * PRE1_N + col] = (h16)fmaxf(acc[r] + bb, 0.0f);
      }
    }
    __syncthreads();

    // ---- phase 2: prenet layer 2 (K=256, N=128) ----
    if (wv < 8) {
      const int j = wv * 16;
      v8f acc = z8();
      acc = wmma_acc_seg(acc, S.pre1, PRE1_N, W2, PRE1_N, j, 0, 8);
      const int col = j + (lane & 15);
      const float bb = b2[col];
#pragma unroll
      for (int r = 0; r < 8; ++r) {
        int m = r + ((lane >> 4) << 3);
        S.x2[m * PRE2_N + col] = (h16)fmaxf(acc[r] + bb, 0.0f);
      }
    }
    __syncthreads();

    // ---- phase 3: attention GRU, input = [x2 | ctx] (K = 128 + 256) ----
    {
      float hnew[8];
      if (wv < 16)
        gru_tile(wv * 16, S.x2, PRE2_N, 4, S.ctx, ENC_D, 8, PRE2_N,
                 WihA, PRE2_N + ENC_D, S.h_att, WhhA, att_bih, att_bhh, hnew);
      __syncthreads();
      if (wv < 16) {
        const int col = wv * 16 + (lane & 15);
#pragma unroll
        for (int r = 0; r < 8; ++r) {
          int m = r + ((lane >> 4) << 3);
          S.h_att[m * RNN + col] = (h16)hnew[r];
        }
      }
      __syncthreads();
    }

    // ---- phase 4: query = att_h @ Wq^T ----
    if (wv < 16) {
      const int j = wv * 16;
      v8f acc = z8();
      acc = wmma_acc_seg(acc, S.h_att, RNN, Wq, RNN, j, 0, 8);
      const int col = j + (lane & 15);
#pragma unroll
      for (int r = 0; r < 8; ++r) {
        int m = r + ((lane >> 4) << 3);
        S.q[m * RNN + col] = acc[r];
      }
    }
    __syncthreads();

    // ---- phase 5: scores[m][tt] = v . tanh(q + pm) (masked), all 32 waves ----
    for (int idx = tid; idx < 16 * T_ENC; idx += NTHR) {
      int m = idx / T_ENC, tt = idx - m * T_ENC;
      float s = -3.4e38f;
      if (tt < S.mlen[m]) {
        const float* prow = pm + ((size_t)(b0 + m) * T_ENC + tt) * ENC_D;
        const float* qrow = &S.q[m * RNN];
        s = 0.0f;
        for (int c = 0; c < ENC_D; ++c)
          s += S.vvec[c] * fast_tanh(qrow[c] + prow[c]);
      }
      S.scores[m * T_ENC + tt] = s;
    }
    __syncthreads();

    // ---- phase 6: masked softmax, one wave per row; write aligns out ----
    if (wv < 16) {
      const int row = wv;
      const int L = S.mlen[row];
      float mx = -3.4e38f;
      for (int tt = lane; tt < T_ENC; tt += 32)
        mx = fmaxf(mx, S.scores[row * T_ENC + tt]);
      mx = wred_max(mx);
      float sum = 0.0f;
      for (int tt = lane; tt < T_ENC; tt += 32) {
        float e = (tt < L) ? __expf(S.scores[row * T_ENC + tt] - mx) : 0.0f;
        S.scores[row * T_ENC + tt] = e;
        sum += e;
      }
      sum = wred_sum(sum);
      const float inv = 1.0f / sum;
      float* ao = out_align + ((size_t)(b0 + row) * T_DEC + t) * T_ENC;
      for (int tt = lane; tt < T_ENC; tt += 32) {
        float a = S.scores[row * T_ENC + tt] * inv;
        S.scores[row * T_ENC + tt] = a;
        ao[tt] = a;
      }
    }
    __syncthreads();

    // ---- phase 7: ctx = align @ enc (per-row B => VALU), all 32 waves ----
    for (int idx = tid; idx < 16 * ENC_D; idx += NTHR) {
      int m = idx >> 8, c = idx & 255;
      const int L = S.mlen[m];
      float s = 0.0f;
      const float* erow = enc + ((size_t)(b0 + m) * T_ENC) * ENC_D + c;
      for (int tt = 0; tt < L; ++tt)
        s += S.scores[m * T_ENC + tt] * erow[(size_t)tt * ENC_D];
      S.ctx[m * ENC_D + c] = (h16)s;
    }
    __syncthreads();

    // ---- phase 8: d = [att_h | ctx] @ Wproj^T + proj_b ----
    if (wv < 16) {
      const int j = wv * 16;
      v8f acc = z8();
      acc = wmma_acc_seg(acc, S.h_att, RNN, Wproj, 512, j, 0, 8);
      acc = wmma_acc_seg(acc, S.ctx, ENC_D, Wproj, 512, j, 256, 8);
      const int col = j + (lane & 15);
      const float bb = proj_b[col];
#pragma unroll
      for (int r = 0; r < 8; ++r) {
        int m = r + ((lane >> 4) << 3);
        float v = acc[r] + bb;
        S.d[m * RNN + col] = v;
        S.df16[m * RNN + col] = (h16)v;
      }
    }
    __syncthreads();

    // ---- phase 9: decoder GRU 0 + residual ----
    {
      float hnew[8], dnew[8];
      if (wv < 16) {
        gru_tile(wv * 16, S.df16, RNN, 8, (const h16*)0, 0, 0, 0,
                 Wih0, RNN, S.h1, Whh0, d0_bih, d0_bhh, hnew);
        const int col = wv * 16 + (lane & 15);
#pragma unroll
        for (int r = 0; r < 8; ++r) {
          int m = r + ((lane >> 4) << 3);
          dnew[r] = hnew[r] + S.d[m * RNN + col];
        }
      }
      __syncthreads();
      if (wv < 16) {
        const int col = wv * 16 + (lane & 15);
#pragma unroll
        for (int r = 0; r < 8; ++r) {
          int m = r + ((lane >> 4) << 3);
          S.h1[m * RNN + col] = (h16)hnew[r];
          S.d[m * RNN + col] = dnew[r];
          S.df16[m * RNN + col] = (h16)dnew[r];
        }
      }
      __syncthreads();
    }

    // ---- phase 10: decoder GRU 1 + residual ----
    {
      float hnew[8], dnew[8];
      if (wv < 16) {
        gru_tile(wv * 16, S.df16, RNN, 8, (const h16*)0, 0, 0, 0,
                 Wih1, RNN, S.h2, Whh1, d1_bih, d1_bhh, hnew);
        const int col = wv * 16 + (lane & 15);
#pragma unroll
        for (int r = 0; r < 8; ++r) {
          int m = r + ((lane >> 4) << 3);
          dnew[r] = hnew[r] + S.d[m * RNN + col];
        }
      }
      __syncthreads();
      if (wv < 16) {
        const int col = wv * 16 + (lane & 15);
#pragma unroll
        for (int r = 0; r < 8; ++r) {
          int m = r + ((lane >> 4) << 3);
          S.h2[m * RNN + col] = (h16)hnew[r];
          S.d[m * RNN + col] = dnew[r];
          S.df16[m * RNN + col] = (h16)dnew[r];
        }
      }
      __syncthreads();
    }

    // ---- phase 11: mel = [d | ctx] @ Wmel[0:80]^T + mel_b ----
    if (wv < 5) {
      const int j = wv * 16;
      v8f acc = z8();
      acc = wmma_acc_seg(acc, S.df16, RNN, Wmel, 512, j, 0, 8);
      acc = wmma_acc_seg(acc, S.ctx, ENC_D, Wmel, 512, j, 256, 8);
      const int col = j + (lane & 15);
      const float bb = mel_b[col];
#pragma unroll
      for (int r = 0; r < 8; ++r) {
        int m = r + ((lane >> 4) << 3);
        out_mel[((size_t)(b0 + m) * T_DEC + t) * MEL_N + col] = acc[r] + bb;
      }
    }

    // ---- phase 12: stop token ----
    if (tid < 16) {
      const int m = tid;
      float s = stop_b[0];
      for (int c = 0; c < RNN; ++c) s += (float)S.df16[m * RNN + c] * stop_w[c];
      for (int c = 0; c < ENC_D; ++c)
        s += (float)S.ctx[m * ENC_D + c] * stop_w[RNN + c];
      out_stop[(size_t)(b0 + m) * T_DEC + t] = sigm(s);
    }
    __syncthreads();
  }
}

// ---------------------------------------------------------------------------
// host launcher
// ---------------------------------------------------------------------------
extern "C" void kernel_launch(void* const* d_in, const int* in_sizes, int n_in,
                              void* d_out, int out_size, void* d_ws, size_t ws_size,
                              hipStream_t stream) {
  const float* enc      = (const float*)d_in[0];
  const float* inputs   = (const float*)d_in[1];
  const int*   memlen   = (const int*)d_in[2];
  const float* pre_w1   = (const float*)d_in[3];
  const float* pre_b1   = (const float*)d_in[4];
  const float* pre_w2   = (const float*)d_in[5];
  const float* pre_b2   = (const float*)d_in[6];
  const float* mem_w    = (const float*)d_in[7];
  const float* att_wih  = (const float*)d_in[8];
  const float* att_whh  = (const float*)d_in[9];
  const float* att_bih  = (const float*)d_in[10];
  const float* att_bhh  = (const float*)d_in[11];
  const float* query_w  = (const float*)d_in[12];
  const float* v_w      = (const float*)d_in[13];
  const float* proj_w   = (const float*)d_in[14];
  const float* proj_b   = (const float*)d_in[15];
  const float* d0_wih   = (const float*)d_in[16];
  const float* d0_whh   = (const float*)d_in[17];
  const float* d0_bih   = (const float*)d_in[18];
  const float* d0_bhh   = (const float*)d_in[19];
  const float* d1_wih   = (const float*)d_in[20];
  const float* d1_whh   = (const float*)d_in[21];
  const float* d1_bih   = (const float*)d_in[22];
  const float* d1_bhh   = (const float*)d_in[23];
  const float* mel_w    = (const float*)d_in[24];
  const float* mel_b    = (const float*)d_in[25];
  const float* stop_w   = (const float*)d_in[26];
  const float* stop_b   = (const float*)d_in[27];
  (void)in_sizes; (void)n_in; (void)out_size; (void)ws_size;

  char* ws = (char*)d_ws;
  size_t off = 0;
  auto alloc_bytes = [&](size_t nbytes) -> void* {
    void* p = (void*)(ws + off);
    off += ((nbytes + 255) / 256) * 256;
    return p;
  };
  float* pm  = (float*)alloc_bytes((size_t)B_TOT * T_ENC * ENC_D * sizeof(float));
  auto alloc_h16 = [&](size_t n) -> h16* { return (h16*)alloc_bytes(n * sizeof(h16)); };
  h16* W1p   = alloc_h16((size_t)PRE1_N * 96);
  h16* W2c   = alloc_h16((size_t)PRE2_N * PRE1_N);
  h16* Wqc   = alloc_h16((size_t)RNN * RNN);
  h16* Wmc   = alloc_h16((size_t)RNN * ENC_D);
  h16* WihA  = alloc_h16((size_t)3 * RNN * (PRE2_N + ENC_D));
  h16* WhhA  = alloc_h16((size_t)3 * RNN * RNN);
  h16* Wpr   = alloc_h16((size_t)RNN * 512);
  h16* Wih0c = alloc_h16((size_t)3 * RNN * RNN);
  h16* Whh0c = alloc_h16((size_t)3 * RNN * RNN);
  h16* Wih1c = alloc_h16((size_t)3 * RNN * RNN);
  h16* Whh1c = alloc_h16((size_t)3 * RNN * RNN);
  h16* Wmelc = alloc_h16((size_t)MEL_N * 512);

  auto conv = [&](const float* s, h16* d, int n, int ks, int kp) {
    int tot = n * kp;
    convert_w_kernel<<<(tot + 255) / 256, 256, 0, stream>>>(s, d, n, ks, kp);
  };
  conv(pre_w1,  W1p,   PRE1_N, MEL_N, 96);
  conv(pre_w2,  W2c,   PRE2_N, PRE1_N, PRE1_N);
  conv(query_w, Wqc,   RNN, RNN, RNN);
  conv(mem_w,   Wmc,   RNN, ENC_D, ENC_D);
  conv(att_wih, WihA,  3 * RNN, PRE2_N + ENC_D, PRE2_N + ENC_D);
  conv(att_whh, WhhA,  3 * RNN, RNN, RNN);
  conv(proj_w,  Wpr,   RNN, 512, 512);
  conv(d0_wih,  Wih0c, 3 * RNN, RNN, RNN);
  conv(d0_whh,  Whh0c, 3 * RNN, RNN, RNN);
  conv(d1_wih,  Wih1c, 3 * RNN, RNN, RNN);
  conv(d1_whh,  Whh1c, 3 * RNN, RNN, RNN);
  conv(mel_w,   Wmelc, MEL_N, 512, 512);  // only first 80 rows of mel_w used

  pm_kernel<<<(B_TOT * T_ENC) / 16, 512, 0, stream>>>(enc, Wmc, pm);

  float* out_mel   = (float*)d_out;
  float* out_stop  = out_mel + (size_t)B_TOT * T_DEC * MEL_N;
  float* out_align = out_stop + (size_t)B_TOT * T_DEC;

  decoder_kernel<<<B_TOT / 16, NTHR, sizeof(SMem), stream>>>(
      enc, inputs, memlen, pre_b1, pre_b2, att_bih, att_bhh, v_w, proj_b,
      d0_bih, d0_bhh, d1_bih, d1_bhh, mel_b, stop_w, stop_b,
      pm, W1p, W2c, Wqc, WihA, WhhA, Wpr, Wih0c, Whh0c, Wih1c, Whh1c, Wmelc,
      out_mel, out_stop, out_align);
}
